// TGATTGM_13202729467940
// MI455X (gfx1250) — compile-verified, matrix operand order checked by reading, not run on hardware
//
#include <hip/hip_runtime.h>
#include <hip/hip_bf16.h>
#include <math.h>

// ---------------- CDNA5 WMMA types ----------------
typedef __attribute__((ext_vector_type(16))) __bf16 v16bf;
typedef __attribute__((ext_vector_type(8)))  float  v8f;
typedef __attribute__((ext_vector_type(4)))  unsigned int v4u;
typedef __attribute__((ext_vector_type(8)))  unsigned int v8u;

__device__ __forceinline__ unsigned short f2bf(float f) {
    unsigned int u = __builtin_bit_cast(unsigned int, f);
    unsigned int r = u + 0x7FFFu + ((u >> 16) & 1u);   // round-to-nearest-even
    return (unsigned short)(r >> 16);
}

// ---------------- Generic bf16 WMMA GEMM ----------------
// C[M][gridDim.x*16] = A[M][KD] * Bt[gridDim.x*16][KD]^T (+bias, opt relu)
// A row-major (lda); Bt is output-column-major (row n = weights of output col n).
// KD: compile-time, multiple of 32. One wave per 16x16 tile.
// Out-of-range M rows: row pointer clamped (A row m only feeds C row m, which
// is never stored), so the hot loop has NO predication: 4x b128 loads + 1 wmma.
template<int KD>
__global__ void tg_gemm_bf16(const unsigned short* __restrict__ A, int lda,
                             const unsigned short* __restrict__ Bt, int ldb,
                             const float* __restrict__ bias, int Nbias,
                             void* __restrict__ Cout, int ldc,
                             int M, int relu, int out_bf16)
{
    int tileN = blockIdx.x, tileM = blockIdx.y;
    int lane = threadIdx.x & 31;
    int lr = lane & 15;        // row/col-within-tile selector
    int lh = lane >> 4;        // half selector (K 0..7/16..23 vs 8..15/24..31)
    int m = tileM * 16 + lr;
    int n = tileN * 16 + lr;
    int mc = (m < M) ? m : (M - 1);        // clamp: safe, never stored
    const unsigned short* arow = A + (size_t)mc * lda + lh * 8;
    const unsigned short* brow = Bt + (size_t)n * ldb + lh * 8;

    v8f acc = {};
#pragma unroll
    for (int kk = 0; kk < KD; kk += 32) {
        v4u a0 = *(const v4u*)(arow + kk);
        v4u a1 = *(const v4u*)(arow + kk + 16);
        v4u b0 = *(const v4u*)(brow + kk);
        v4u b1 = *(const v4u*)(brow + kk + 16);
        v8u au, bu;
        au[0] = a0[0]; au[1] = a0[1]; au[2] = a0[2]; au[3] = a0[3];
        au[4] = a1[0]; au[5] = a1[1]; au[6] = a1[2]; au[7] = a1[3];
        bu[0] = b0[0]; bu[1] = b0[1]; bu[2] = b0[2]; bu[3] = b0[3];
        bu[4] = b1[0]; bu[5] = b1[1]; bu[6] = b1[2]; bu[7] = b1[3];
        v16bf af = __builtin_bit_cast(v16bf, au);
        v16bf bf = __builtin_bit_cast(v16bf, bu);
        acc = __builtin_amdgcn_wmma_f32_16x16x32_bf16(false, af, false, bf,
                                                      (short)0, acc, false, false);
    }

    float bn = (bias && n < Nbias) ? bias[n] : 0.0f;
#pragma unroll
    for (int r = 0; r < 8; ++r) {
        int mr = tileM * 16 + r + lh * 8;   // VGPR r: lanes0-15 -> M=r, lanes16-31 -> M=r+8
        if (mr < M) {
            float v = acc[r] + bn;
            if (relu) v = fmaxf(v, 0.0f);
            if (out_bf16) ((unsigned short*)Cout)[(size_t)mr * ldc + n] = f2bf(v);
            else          ((float*)Cout)[(size_t)mr * ldc + n] = v;
        }
    }
}

// ---------------- Weight prep: pad + transpose to bf16 ----------------
__global__ void tg_prep_weights(int l,
                                const float* __restrict__ wq, const float* __restrict__ wk,
                                const float* __restrict__ wv, const float* __restrict__ f1w,
                                const float* __restrict__ f2w,
                                unsigned short* WqT, unsigned short* WkT,
                                unsigned short* WvT, unsigned short* F1T,
                                unsigned short* F2T)
{
    int tid = blockIdx.x * blockDim.x + threadIdx.x;
    int nt = gridDim.x * blockDim.x;
    // WqT [176][288] : out col c, k over 272 (q_in)
    for (int i = tid; i < 176 * 288; i += nt) {
        int c = i / 288, k = i % 288;
        WqT[i] = (c < 172 && k < 272) ? f2bf(wq[((size_t)l * 272 + k) * 172 + c]) : 0;
    }
    // WkT [2][448][96] : p_h[j] = sum_d q_h[d] * wk[j][h*86+d]  (out col j, K=d)
    for (int i = tid; i < 2 * 448 * 96; i += nt) {
        int h = i / (448 * 96); int r = i % (448 * 96); int j = r / 96, d = r % 96;
        WkT[i] = (j < 444 && d < 86) ? f2bf(wk[((size_t)l * 444 + j) * 172 + h * 86 + d]) : 0;
    }
    // WvT [2][96][448] : o_h[d] = sum_j m_h[j] * wv[j][h*86+d]  (out col d, K=j)
    for (int i = tid; i < 2 * 96 * 448; i += nt) {
        int h = i / (96 * 448); int r = i % (96 * 448); int d = r / 448, j = r % 448;
        WvT[i] = (d < 86 && j < 444) ? f2bf(wv[((size_t)l * 444 + j) * 172 + h * 86 + d]) : 0;
    }
    // F1T [192][352]
    for (int i = tid; i < 192 * 352; i += nt) {
        int c = i / 352, k = i % 352;
        F1T[i] = (c < 172 && k < 344) ? f2bf(f1w[((size_t)l * 344 + k) * 172 + c]) : 0;
    }
    // F2T [176][192]
    for (int i = tid; i < 176 * 192; i += nt) {
        int c = i / 192, k = i % 192;
        F2T[i] = (c < 172 && k < 172) ? f2bf(f2w[((size_t)l * 172 + k) * 172 + c]) : 0;
    }
}

// ---------------- q_in builder: [feat(172) | cos(time_b)(100) | 0-pad] ----------------
__global__ void tg_build_qin(int n, const int* __restrict__ ids,
                             const float* __restrict__ sf,
                             const float* __restrict__ time_b,
                             unsigned short* __restrict__ qin)
{
    int i = blockIdx.x; if (i >= n) return;
    int gid = ids[i];
    unsigned short* row = qin + (size_t)i * 288;
    for (int c = threadIdx.x; c < 288; c += blockDim.x) {
        float v;
        if (c < 172)      v = sf[(size_t)gid * 172 + c];
        else if (c < 272) v = cosf(time_b[c - 172]);   // time_enc(0)
        else              v = 0.0f;
        row[c] = f2bf(v);
    }
}

// ---------------- split q into padded per-head bf16 + q.bk correction ----------------
__global__ void tg_split_q(int n, const float* __restrict__ qf,
                           const float* __restrict__ bk_l,
                           unsigned short* __restrict__ Qh, float* __restrict__ qbk)
{
    int i = blockIdx.x; if (i >= n) return;
    int tid = threadIdx.x;
    for (int c = tid; c < 192; c += 128) {
        int h = c / 96, d = c % 96;
        float v = (d < 86) ? qf[(size_t)i * 176 + h * 86 + d] : 0.0f;
        Qh[(size_t)i * 192 + c] = f2bf(v);
    }
    int wave = tid >> 5, lane = tid & 31;
    if (wave < 2) {
        float s = 0.0f;
        for (int d = lane; d < 86; d += 32)
            s += qf[(size_t)i * 176 + wave * 86 + d] * bk_l[wave * 86 + d];
        for (int off = 16; off; off >>= 1) s += __shfl_down(s, off, 32);
        if (lane == 0) qbk[i * 2 + wave] = s;
    }
}

// ---------------- fused per-node attention: scores, softmax, m = sum a*kv ----------------
__global__ void tg_attention(int n, int use_z,
                             const int* __restrict__ nbr, const unsigned char* __restrict__ mask,
                             const float* __restrict__ times, const float* __restrict__ nbr_times,
                             const float* __restrict__ edge, const float* __restrict__ sf,
                             const float* __restrict__ z, const int* __restrict__ g2l,
                             const float* __restrict__ P, const float* __restrict__ qbk,
                             const float* __restrict__ time_w, const float* __restrict__ time_b,
                             unsigned short* __restrict__ Mout, float* __restrict__ suma)
{
    int i = blockIdx.x; if (i >= n) return;
    int tid = threadIdx.x;
    __shared__ float p_sh[2 * 448];
    __shared__ float rows[20][448];
    __shared__ float scv[20][2];

    for (int c = tid; c < 896; c += 128) p_sh[c] = P[(size_t)i * 896 + c];
    float ti = times[i];
    for (int j = 0; j < 20; ++j) {
        int e0 = i * 20 + j;
        int gid = nbr[e0];
        unsigned char mk = mask[e0];
        for (int c = tid; c < 172; c += 128) {
            float f = (use_z && mk) ? z[(size_t)g2l[gid] * 172 + c]
                                    : sf[(size_t)gid * 172 + c];
            rows[j][c] = f;
        }
        for (int c = tid; c < 172; c += 128)
            rows[j][172 + c] = edge[(size_t)e0 * 172 + c];
        float dt = mk ? (ti - nbr_times[e0]) : 0.0f;
        for (int c = tid; c < 100; c += 128)
            rows[j][344 + c] = cosf(dt * time_w[c] + time_b[c]);
        if (tid < 4) rows[j][444 + tid] = 0.0f;
    }
    __syncthreads();

    int wave = tid >> 5, lane = tid & 31;
    for (int idx = wave; idx < 40; idx += 4) {   // 20 neighbors x 2 heads
        int j = idx >> 1, h = idx & 1;
        float s = 0.0f;
        for (int c = lane; c < 448; c += 32) s += rows[j][c] * p_sh[h * 448 + c];
        for (int off = 16; off; off >>= 1) s += __shfl_down(s, off, 32);
        if (lane == 0) scv[j][h] = s;
    }
    __syncthreads();

    if (tid < 2) {   // softmax per head (K=20, trivial)
        int h = tid;
        float qb = qbk[i * 2 + h];
        const float scale = 0.10783277320343841f;   // 1/sqrt(86)
        float sc[20]; float mx = -1e30f;
        for (int j = 0; j < 20; ++j) {
            unsigned char mk = mask[i * 20 + j];
            float v = mk ? (scv[j][h] + qb) * scale : -1e10f;
            sc[j] = v; mx = fmaxf(mx, v);
        }
        float se = 0.0f;
        for (int j = 0; j < 20; ++j) { sc[j] = expf(sc[j] - mx); se += sc[j]; }
        float inv = 1.0f / se; float sa = 0.0f;
        for (int j = 0; j < 20; ++j) {
            unsigned char mk = mask[i * 20 + j];
            float a = mk ? sc[j] * inv : 0.0f;
            scv[j][h] = a; sa += a;
        }
        suma[i * 2 + h] = sa;
    }
    __syncthreads();

    for (int idx = tid; idx < 896; idx += 128) {   // m_h = sum_j a_jh * kv_row_j
        int h = idx / 448, c = idx % 448;
        float s = 0.0f;
#pragma unroll
        for (int j = 0; j < 20; ++j) s += scv[j][h] * rows[j][c];
        Mout[(size_t)i * 896 + h * 448 + c] = f2bf(s);
    }
}

// ---------------- pack FFN input: [o + suma*bv (172) | node_feat (172) | pad] ----------------
__global__ void tg_pack_f1in(int n, const int* __restrict__ ids,
                             const float* __restrict__ Oh, const float* __restrict__ suma,
                             const float* __restrict__ bv_l, const float* __restrict__ sf,
                             unsigned short* __restrict__ f1in)
{
    int i = blockIdx.x; if (i >= n) return;
    int gid = ids[i];
    for (int c = threadIdx.x; c < 352; c += 128) {
        float v;
        if (c < 172) {
            int h = c / 86, d = c % 86;
            v = Oh[(size_t)i * 192 + h * 96 + d] + suma[i * 2 + h] * bv_l[c];
        } else if (c < 344) {
            v = sf[(size_t)gid * 172 + (c - 172)];
        } else v = 0.0f;
        f1in[(size_t)i * 352 + c] = f2bf(v);
    }
}

__global__ void tg_scatter(int n, const int* __restrict__ ids, const int* __restrict__ g2l,
                           const float* __restrict__ outb, float* __restrict__ zout)
{
    int i = blockIdx.x; if (i >= n) return;
    int row = g2l[ids[i]];
    for (int c = threadIdx.x; c < 172; c += blockDim.x)
        zout[(size_t)row * 172 + c] = outb[(size_t)i * 176 + c];
}

// ---------------- host ----------------
extern "C" void kernel_launch(void* const* d_in, const int* in_sizes, int n_in,
                              void* d_out, int out_size, void* d_ws, size_t ws_size,
                              hipStream_t stream)
{
    (void)in_sizes; (void)n_in; (void)out_size; (void)ws_size;
    const float* sf         = (const float*)d_in[0];
    const float* time_w     = (const float*)d_in[1];
    const float* time_b     = (const float*)d_in[2];
    const float* wq         = (const float*)d_in[3];
    const float* bq         = (const float*)d_in[4];
    const float* wk         = (const float*)d_in[5];
    const float* bk         = (const float*)d_in[6];
    const float* wv         = (const float*)d_in[7];
    const float* bv         = (const float*)d_in[8];
    const float* f1w        = (const float*)d_in[9];
    const float* f1b        = (const float*)d_in[10];
    const float* f2w        = (const float*)d_in[11];
    const float* f2b        = (const float*)d_in[12];
    const float* times0     = (const float*)d_in[13];
    const float* nbr_times0 = (const float*)d_in[14];
    const float* times1     = (const float*)d_in[15];
    const float* nbr_times1 = (const float*)d_in[16];
    const float* edge0      = (const float*)d_in[17];
    const float* edge1      = (const float*)d_in[18];
    const int*   nids0      = (const int*)d_in[19];
    const int*   nbr0       = (const int*)d_in[20];
    const int*   nbr1       = (const int*)d_in[21];
    const int*   g2l        = (const int*)d_in[22];
    const unsigned char* mask0 = (const unsigned char*)d_in[23];
    const unsigned char* mask1 = (const unsigned char*)d_in[24];
    float* zout = (float*)d_out;

    const int N1 = 10240, N0 = 512;

    char* ws = (char*)d_ws;
    size_t off = 0;
    auto alloc = [&](size_t bytes) -> void* {
        void* p = ws + off; off = (off + bytes + 255) & ~(size_t)255; return p;
    };
    unsigned short* WqT   = (unsigned short*)alloc((size_t)176 * 288 * 2);
    unsigned short* WkT   = (unsigned short*)alloc((size_t)2 * 448 * 96 * 2);
    unsigned short* WvT   = (unsigned short*)alloc((size_t)2 * 96 * 448 * 2);
    unsigned short* F1T   = (unsigned short*)alloc((size_t)192 * 352 * 2);
    unsigned short* F2T   = (unsigned short*)alloc((size_t)176 * 192 * 2);
    unsigned short* qin   = (unsigned short*)alloc((size_t)N1 * 288 * 2);
    float*          qf    = (float*)alloc((size_t)N1 * 176 * 4);
    unsigned short* Qh    = (unsigned short*)alloc((size_t)N1 * 192 * 2);
    float*          qbk   = (float*)alloc((size_t)N1 * 2 * 4);
    float*          Pbuf  = (float*)alloc((size_t)N1 * 896 * 4);
    unsigned short* Mbuf  = (unsigned short*)alloc((size_t)N1 * 896 * 2);
    float*          suma  = (float*)alloc((size_t)N1 * 2 * 4);
    float*          Ohbuf = (float*)alloc((size_t)N1 * 192 * 4);
    unsigned short* f1in  = (unsigned short*)alloc((size_t)N1 * 352 * 2);
    unsigned short* hbuf  = (unsigned short*)alloc((size_t)N1 * 192 * 2);
    float*          outbuf= (float*)alloc((size_t)N1 * 176 * 4);

    auto run_layer = [&](int l, int n, const int* ids, const int* nbr,
                         const unsigned char* mask, const float* times,
                         const float* nbr_times, const float* edge, int use_z) {
        int mt = (n + 15) / 16;
        tg_prep_weights<<<64, 256, 0, stream>>>(l, wq, wk, wv, f1w, f2w,
                                                WqT, WkT, WvT, F1T, F2T);
        tg_build_qin<<<n, 128, 0, stream>>>(n, ids, sf, time_b, qin);
        // q = q_in @ Wq + bq   -> f32 [n][176]
        tg_gemm_bf16<288><<<dim3(11, mt), 32, 0, stream>>>(qin, 288, WqT, 288,
                                                           bq + l * 172, 172,
                                                           (void*)qf, 176, n, 0, 0);
        tg_split_q<<<n, 128, 0, stream>>>(n, qf, bk + l * 172, Qh, qbk);
        // p_h = q_h @ Wk_h^T   -> f32 [n][2][448]
        for (int h = 0; h < 2; ++h)
            tg_gemm_bf16<96><<<dim3(28, mt), 32, 0, stream>>>(Qh + h * 96, 192,
                                                              WkT + (size_t)h * 448 * 96, 96,
                                                              nullptr, 0,
                                                              (void*)(Pbuf + h * 448), 896,
                                                              n, 0, 0);
        tg_attention<<<n, 128, 0, stream>>>(n, use_z, nbr, mask, times, nbr_times,
                                            edge, sf, zout, g2l, Pbuf, qbk,
                                            time_w, time_b, Mbuf, suma);
        // o_h = m_h @ Wv_h   -> f32 [n][2][96]
        for (int h = 0; h < 2; ++h)
            tg_gemm_bf16<448><<<dim3(6, mt), 32, 0, stream>>>(Mbuf + h * 448, 896,
                                                              WvT + (size_t)h * 96 * 448, 448,
                                                              nullptr, 0,
                                                              (void*)(Ohbuf + h * 96), 192,
                                                              n, 0, 0);
        tg_pack_f1in<<<n, 128, 0, stream>>>(n, ids, Ohbuf, suma, bv + l * 172, sf, f1in);
        // h = relu(f1in @ F1 + f1b) -> bf16 [n][192] (pad cols zero)
        tg_gemm_bf16<352><<<dim3(12, mt), 32, 0, stream>>>(f1in, 352, F1T, 352,
                                                           f1b + l * 172, 172,
                                                           (void*)hbuf, 192, n, 1, 1);
        // out = h @ F2 + f2b -> f32 [n][176]
        tg_gemm_bf16<192><<<dim3(11, mt), 32, 0, stream>>>(hbuf, 192, F2T, 192,
                                                           f2b + l * 172, 172,
                                                           (void*)outbuf, 176, n, 0, 0);
        tg_scatter<<<n, 128, 0, stream>>>(n, ids, g2l, outbuf, zout);
    };

    // Layer 1: nodes = nbr0 flattened (10240), neighbors = nbr1
    run_layer(1, N1, nbr0, nbr1, mask1, times1, nbr_times1, edge1, /*use_z=*/0);
    // Layer 0: nodes = nids0 (512), neighbors = nbr0, kv feat from z when masked
    run_layer(0, N0, nids0, nbr0, mask0, times0, nbr_times0, edge0, /*use_z=*/1);
}
